// CRCEncoder_24077586662020
// MI455X (gfx1250) — matrix operand order repaired and checked by reading, not compiled, not run on hardware
//
#include <hip/hip_runtime.h>

typedef __attribute__((ext_vector_type(16))) _Float16 v16h;
typedef __attribute__((ext_vector_type(8)))  float    v8f;
typedef __attribute__((ext_vector_type(4)))  float    v4f;

#define BATCH   16384
#define KDIM    4096
#define NCRC    24
#define OUTW    (KDIM + NCRC)        // 4120
#define KSPLIT  4
#define KPW     (KDIM / KSPLIT)      // 1024 K-values per wave
#define KSTEPS_TOTAL (KDIM / 32)     // 128 total 32-wide K steps
#define ROWBLKS (BATCH / 16)         // 1024 row blocks of 16

// ---------------------------------------------------------------------------
// Kernel 1: zero the 24 parity columns (they serve as f32 accumulators).
// ---------------------------------------------------------------------------
__global__ void zero_parity(float* __restrict__ out) {
  int idx = blockIdx.x * blockDim.x + threadIdx.x;
  if (idx >= BATCH * NCRC) return;
  int row = idx / NCRC, col = idx - row * NCRC;
  out[(size_t)row * OUTW + KDIM + col] = 0.0f;
}

// ---------------------------------------------------------------------------
// Kernel 2: pack g_mat (4096x24 f32 row-major) into CDNA5 WMMA B fragments.
// Fragment (kstep, ntile, lane) = 16 f16 values, contiguous 32 bytes.
// ISA B layout (16-bit, 32x16): lane%16 selects column N; lanes 0-15 hold
// K {0..7,16..23} of the 32-K step, lanes 16-31 hold K {8..15,24..31} (2 halves per VGPR).
// Columns 24..31 are zero padding.
// ---------------------------------------------------------------------------
__global__ void pack_b(const float* __restrict__ g, _Float16* __restrict__ bp) {
  int tid = blockIdx.x * blockDim.x + threadIdx.x;
  if (tid >= KSTEPS_TOTAL * 2 * 32) return;
  int lane  = tid & 31;
  int ntile = (tid >> 5) & 1;
  int kstep = tid >> 6;
  int n   = ntile * 16 + (lane & 15);
  int kk0 = (lane < 16) ? 0 : 8;
  _Float16* dst = bp + (size_t)tid * 16;
  for (int i = 0; i < 16; ++i) {
    int k = kstep * 32 + kk0 + (i < 8 ? i : i + 8);
    float v = (n < NCRC) ? g[(size_t)k * NCRC + n] : 0.0f;
    dst[i] = (_Float16)v;
  }
}

// ---------------------------------------------------------------------------
// Kernel 3: fused streaming copy + WMMA parity GEMM.
// Each wave: one 16-row block x one K-split of 1024. Per 32-K step it loads
// the bits tile in exact A-fragment order (lanes 0-15: K{0..7,16..23},
// lanes 16-31: K{8..15,24..31}), stores it to the concat region, converts to
// f16 and issues two v_wmma_f32_16x16x32_f16 (N tiles 0-15, 16-31).
// Streaming traffic (input read-once, copy write-once) is marked
// non-temporal so the 192 MB L2 stays free for the reused B fragments and
// the atomic parity accumulators. Partial sums are merged across K-splits
// with exact f32 atomic adds (integer-valued -> order independent).
// ---------------------------------------------------------------------------
__global__ void crc_wmma(const float* __restrict__ in,
                         const _Float16* __restrict__ bp,
                         float* __restrict__ out) {
  int wave   = (blockIdx.x * blockDim.x + threadIdx.x) >> 5;  // wave32
  int lane   = threadIdx.x & 31;
  int ksplit = wave & (KSPLIT - 1);
  int rowblk = wave >> 2;                     // 0..1023
  size_t r0  = (size_t)rowblk * 16;
  int kbase0 = ksplit * KPW;

  int mrow   = lane & 15;
  int laneHi = lane >> 4;                     // 0: lanes 0-15, 1: lanes 16-31
  size_t row = r0 + mrow;

  const float* inrow  = in  + row * (size_t)KDIM;
  float*       cpyrow = out + row * (size_t)OUTW;

  v8f acc0 = {}; v8f acc1 = {};

  for (int ks = 0; ks < KPW / 32; ++ks) {
    int kbase = kbase0 + ks * 32;
    int c0    = kbase + laneHi * 8;           // A-fragment run 0; run 1 = c0+16

    const v4f* p0 = (const v4f*)(inrow + c0);
    const v4f* p1 = (const v4f*)(inrow + c0 + 16);
    v4f x0 = __builtin_nontemporal_load(p0 + 0);
    v4f x1 = __builtin_nontemporal_load(p0 + 1);
    v4f y0 = __builtin_nontemporal_load(p1 + 0);
    v4f y1 = __builtin_nontemporal_load(p1 + 1);

    // concat copy (cols match input cols, shifted row stride) -- write-once
    v4f* q0 = (v4f*)(cpyrow + c0);
    v4f* q1 = (v4f*)(cpyrow + c0 + 16);
    __builtin_nontemporal_store(x0, q0 + 0);
    __builtin_nontemporal_store(x1, q0 + 1);
    __builtin_nontemporal_store(y0, q1 + 0);
    __builtin_nontemporal_store(y1, q1 + 1);

    // build A fragment (element e -> VGPR e/2, half e%2)
    v16h a;
    a[0]  = (_Float16)x0.x; a[1]  = (_Float16)x0.y;
    a[2]  = (_Float16)x0.z; a[3]  = (_Float16)x0.w;
    a[4]  = (_Float16)x1.x; a[5]  = (_Float16)x1.y;
    a[6]  = (_Float16)x1.z; a[7]  = (_Float16)x1.w;
    a[8]  = (_Float16)y0.x; a[9]  = (_Float16)y0.y;
    a[10] = (_Float16)y0.z; a[11] = (_Float16)y0.w;
    a[12] = (_Float16)y1.x; a[13] = (_Float16)y1.y;
    a[14] = (_Float16)y1.z; a[15] = (_Float16)y1.w;

    // B fragments (pre-packed, L2-resident, heavily reused -> default RT)
    int kstep = kbase >> 5;
    v16h b0 = *(const v16h*)(bp + ((size_t)(kstep * 2 + 0) * 32 + lane) * 16);
    v16h b1 = *(const v16h*)(bp + ((size_t)(kstep * 2 + 1) * 32 + lane) * 16);

    acc0 = __builtin_amdgcn_wmma_f32_16x16x32_f16(false, a, false, b0,
                                                  (short)0, acc0, false, false);
    acc1 = __builtin_amdgcn_wmma_f32_16x16x32_f16(false, a, false, b1,
                                                  (short)0, acc1, false, false);
  }

  // C/D layout: VGPR v, lanes 0-15: M=v, N=lane; lanes 16-31: M=v+8, N=lane-16
  int n0   = lane & 15;
  int mAdd = laneHi * 8;
  for (int v = 0; v < 8; ++v) {
    float* prow = out + (r0 + v + mAdd) * (size_t)OUTW + KDIM;
    atomicAdd(prow + n0, acc0[v]);            // N tile 0: n = n0 < 24 always
    if (n0 < 8) atomicAdd(prow + 16 + n0, acc1[v]);  // N tile 1: n = 16+n0
  }
}

// ---------------------------------------------------------------------------
// Kernel 4: in-place mod 2 on the parity columns (values are exact integers).
// ---------------------------------------------------------------------------
__global__ void finalize_mod2(float* __restrict__ out) {
  int idx = blockIdx.x * blockDim.x + threadIdx.x;
  if (idx >= BATCH * NCRC) return;
  int row = idx / NCRC, col = idx - row * NCRC;
  float* p = out + (size_t)row * OUTW + KDIM + col;
  int iv = (int)(*p + 0.5f);
  *p = (float)(iv & 1);
}

extern "C" void kernel_launch(void* const* d_in, const int* in_sizes, int n_in,
                              void* d_out, int out_size, void* d_ws, size_t ws_size,
                              hipStream_t stream) {
  const float* bits = (const float*)d_in[0];   // [16384, 4096] f32 (0/1)
  const float* g    = (const float*)d_in[1];   // [4096, 24]   f32 (0/1)
  float*       out  = (float*)d_out;           // [16384, 4120] f32
  _Float16*    bp   = (_Float16*)d_ws;         // 256 KB packed B fragments

  (void)in_sizes; (void)n_in; (void)out_size; (void)ws_size;

  zero_parity<<<(BATCH * NCRC + 255) / 256, 256, 0, stream>>>(out);
  pack_b<<<(KSTEPS_TOTAL * 2 * 32 + 255) / 256, 256, 0, stream>>>(g, bp);

  int waves  = ROWBLKS * KSPLIT;               // 4096 waves
  int blocks = waves / 8;                      // 256 threads = 8 wave32s
  crc_wmma<<<blocks, 256, 0, stream>>>(bits, bp, out);

  finalize_mod2<<<(BATCH * NCRC + 255) / 256, 256, 0, stream>>>(out);
}
